// _LoRA_qkv_timm_eval_61761629716786
// MI455X (gfx1250) — matrix operand order repaired and checked by reading, not compiled, not run on hardware
//
#include <hip/hip_runtime.h>

// ---------------- problem constants (from reference) ----------------
#define DD    768          // model dim
#define RR    16           // lora rank
#define TT    10           // tasks
#define BN    6304         // 32*197 tokens
#define BNP   6336         // padded to multiple of 64 (block M tile)
#define E3D   2304         // 3*D output dim
#define TAUF  0.5f
#define EPSF  1e-8f

// ---------------- bf16 helpers (manual RNE, no scalar __bf16 math) ----------------
typedef __attribute__((ext_vector_type(16))) __bf16 v16bf;
typedef __attribute__((ext_vector_type(8)))  float  v8f;

__device__ __forceinline__ unsigned short f2bf(float f) {
    unsigned u = __builtin_bit_cast(unsigned, f);
    unsigned r = 0x7FFFu + ((u >> 16) & 1u);   // round-to-nearest-even
    return (unsigned short)((u + r) >> 16);
}
__device__ __forceinline__ float bf2f(unsigned short h) {
    unsigned u = ((unsigned)h) << 16;
    return __builtin_bit_cast(float, u);
}

union FragU { v16bf v; uint4 q[2]; };

// A fragment (16x32 bf16): lane half selects K-groups {0..7,16..23} vs {8..15,24..31}
__device__ __forceinline__ v16bf load_fragA(const unsigned short* p) {
    FragU f;
    f.q[0] = *(const uint4*)(p);        // K = kg + 0..7
    f.q[1] = *(const uint4*)(p + 16);   // K = kg + 16..23
    return f.v;
}
// B fragment (32x16 bf16): lane half selects K-groups {0..7,16..23} vs {8..15,24..31}
__device__ __forceinline__ v16bf load_fragB(const unsigned short* p) {
    FragU f;
    f.q[0] = *(const uint4*)(p);        // K = kg + 0..7
    f.q[1] = *(const uint4*)(p + 16);   // K = kg + 16..23
    return f.v;
}

// ---------------- kernel 1: per-task coefficients --------------------------------
// coef[t]      = softmax(logits/tau)[t] / (||A_q_t||_F * ||B_q_t||_F + eps)
// coef[T + t]  = same for v adapters
__global__ __launch_bounds__(256)
void lora_coef_kernel(const float* __restrict__ Aq, const float* __restrict__ Bq,
                      const float* __restrict__ Av, const float* __restrict__ Bv,
                      const float* __restrict__ logits, float* __restrict__ coef) {
    __shared__ float sA[256], sB[256];
    const int b   = blockIdx.x;       // 0..2T-1
    const int t   = b % TT;
    const int isv = b / TT;
    const float* A = isv ? Av : Aq;
    const float* B = isv ? Bv : Bq;
    const int tid = threadIdx.x;
    const int nel = RR * DD;          // 12288 per matrix
    float sa = 0.f, sb = 0.f;
    for (int i = tid; i < nel; i += 256) {
        float va = A[t * nel + i]; sa += va * va;
        float vb = B[t * nel + i]; sb += vb * vb;
    }
    sA[tid] = sa; sB[tid] = sb;
    __syncthreads();
    for (int s = 128; s > 0; s >>= 1) {
        if (tid < s) { sA[tid] += sA[tid + s]; sB[tid] += sB[tid + s]; }
        __syncthreads();
    }
    if (tid == 0) {
        float mx = -1e30f;
        for (int j = 0; j < TT; ++j) mx = fmaxf(mx, logits[j] / TAUF);
        float den = 0.f;
        for (int j = 0; j < TT; ++j) den += expf(logits[j] / TAUF - mx);
        float w = expf(logits[t] / TAUF - mx) / den;
        coef[b] = w / (sqrtf(sA[0]) * sqrtf(sB[0]) + EPSF);
    }
}

// ---------------- kernel 2: x -> padded bf16 hi/lo -------------------------------
__global__ __launch_bounds__(256)
void lora_xsplit_kernel(const float* __restrict__ x,
                        unsigned short* __restrict__ xh,
                        unsigned short* __restrict__ xl) {
    int idx = blockIdx.x * 256 + threadIdx.x;
    if (idx >= BNP * DD) return;
    float v = (idx < BN * DD) ? x[idx] : 0.f;   // zero-pad tail rows
    unsigned short h = f2bf(v);
    xh[idx] = h;
    xl[idx] = f2bf(v - bf2f(h));
}

// ---------------- kernel 3: W_eff = qkv_w (+ folded adapters) -> bf16 hi/lo ------
// W_eff[e,k] = qkv_w[e,k] + (e<D      ? sum_t c_q[t] sum_r Aq[t,r,k]*Bq[t,e,r]    : 0)
//                         + (e>=2D    ? sum_t c_v[t] sum_r Av[t,r,k]*Bv[t,e-2D,r] : 0)
__global__ __launch_bounds__(256)
void lora_weff_kernel(const float* __restrict__ qkv_w,
                      const float* __restrict__ Aq, const float* __restrict__ Bq,
                      const float* __restrict__ Av, const float* __restrict__ Bv,
                      const float* __restrict__ coef,
                      unsigned short* __restrict__ wh,
                      unsigned short* __restrict__ wl) {
    int idx = blockIdx.x * 256 + threadIdx.x;
    if (idx >= E3D * DD) return;
    const int e = idx / DD;
    const int k = idx - e * DD;
    float v = qkv_w[idx];
    if (e < DD) {
        float s = 0.f;
        for (int t = 0; t < TT; ++t) {
            float acc = 0.f;
            #pragma unroll
            for (int r = 0; r < RR; ++r)
                acc += Aq[(t * RR + r) * DD + k] * Bq[(t * DD + e) * RR + r];
            s += coef[t] * acc;
        }
        v += s;
    } else if (e >= 2 * DD) {
        const int d = e - 2 * DD;
        float s = 0.f;
        for (int t = 0; t < TT; ++t) {
            float acc = 0.f;
            #pragma unroll
            for (int r = 0; r < RR; ++r)
                acc += Av[(t * RR + r) * DD + k] * Bv[(t * DD + d) * RR + r];
            s += coef[TT + t] * acc;
        }
        v += s;
    }
    unsigned short h = f2bf(v);
    wh[idx] = h;
    wl[idx] = f2bf(v - bf2f(h));
}

// ---------------- kernel 4: out = x @ W_eff^T + b via bf16x3 WMMA ----------------
// Block: 256 threads = 8 waves arranged 2 (M) x 4 (N). Block tile 64 x 256.
// Wave: 2 M-tiles x 4 N-tiles of 16x16 (32x64 per wave).
// K loop: 24 x 32; 24 WMMAs per wave per K step, 24 b128 loads (1.0 loads/WMMA).
__global__ __launch_bounds__(256)
void lora_qkv_gemm_kernel(const unsigned short* __restrict__ xh,
                          const unsigned short* __restrict__ xl,
                          const unsigned short* __restrict__ wh,
                          const unsigned short* __restrict__ wl,
                          const float* __restrict__ bias,
                          float* __restrict__ out) {
    const int lane = threadIdx.x & 31;
    const int wave = threadIdx.x >> 5;
    const int wm = wave & 1;        // 0..1  (M)
    const int wn = wave >> 1;       // 0..3  (N)
    const int m0 = blockIdx.x * 64 + wm * 32;
    const int n0 = blockIdx.y * 256 + wn * 64;

    const int row = lane & 15;               // A: row in tile; B: column in tile
    const int kgA = (lane >> 4) * 8;         // A K-group base
    const int kgB = (lane >> 4) * 8;         // B K-group base

    const unsigned short* pAh[2]; const unsigned short* pAl[2];
    const unsigned short* pBh[4]; const unsigned short* pBl[4];
    #pragma unroll
    for (int mi = 0; mi < 2; ++mi) {
        size_t off = (size_t)(m0 + mi * 16 + row) * DD + kgA;
        pAh[mi] = xh + off;
        pAl[mi] = xl + off;
    }
    #pragma unroll
    for (int ni = 0; ni < 4; ++ni) {
        size_t off = (size_t)(n0 + ni * 16 + row) * DD + kgB;   // W_eff row = output col
        pBh[ni] = wh + off;
        pBl[ni] = wl + off;
    }

    v8f acc[2][4] = {};

    for (int k0 = 0; k0 < DD; k0 += 32) {
        v16bf ah[2], al[2], bh[4], bl[4];
        #pragma unroll
        for (int mi = 0; mi < 2; ++mi) {
            ah[mi] = load_fragA(pAh[mi] + k0);
            al[mi] = load_fragA(pAl[mi] + k0);
        }
        #pragma unroll
        for (int ni = 0; ni < 4; ++ni) {
            bh[ni] = load_fragB(pBh[ni] + k0);
            bl[ni] = load_fragB(pBl[ni] + k0);
        }
        // bf16x3: hi*hi + hi*lo + lo*hi, f32 accumulate.
        // Loop order (term outer, acc tile inner) keeps consecutive WMMAs on
        // distinct accumulators so the scheduler can chain them back-to-back.
        #pragma unroll
        for (int mi = 0; mi < 2; ++mi)
            #pragma unroll
            for (int ni = 0; ni < 4; ++ni)
                acc[mi][ni] = __builtin_amdgcn_wmma_f32_16x16x32_bf16(
                    false, ah[mi], false, bh[ni], (short)0, acc[mi][ni], false, false);
        #pragma unroll
        for (int mi = 0; mi < 2; ++mi)
            #pragma unroll
            for (int ni = 0; ni < 4; ++ni)
                acc[mi][ni] = __builtin_amdgcn_wmma_f32_16x16x32_bf16(
                    false, ah[mi], false, bl[ni], (short)0, acc[mi][ni], false, false);
        #pragma unroll
        for (int mi = 0; mi < 2; ++mi)
            #pragma unroll
            for (int ni = 0; ni < 4; ++ni)
                acc[mi][ni] = __builtin_amdgcn_wmma_f32_16x16x32_bf16(
                    false, al[mi], false, bh[ni], (short)0, acc[mi][ni], false, false);
    }

    // C/D layout: VGPR v, lanes 0-15 -> M=v, N=lane; lanes 16-31 -> M=v+8, N=lane-16
    const int nlane = lane & 15;
    const int mhalf = (lane >> 4) << 3;
    #pragma unroll
    for (int mi = 0; mi < 2; ++mi) {
        #pragma unroll
        for (int ni = 0; ni < 4; ++ni) {
            const int n = n0 + ni * 16 + nlane;
            const float bv = bias[n];
            const int mbase = m0 + mi * 16 + mhalf;
            #pragma unroll
            for (int v = 0; v < 8; ++v) {
                const int m = mbase + v;
                if (m < BN) out[(size_t)m * E3D + n] = acc[mi][ni][v] + bv;
            }
        }
    }
}

// ---------------- launcher -------------------------------------------------------
extern "C" void kernel_launch(void* const* d_in, const int* in_sizes, int n_in,
                              void* d_out, int out_size, void* d_ws, size_t ws_size,
                              hipStream_t stream) {
    (void)in_sizes; (void)n_in; (void)out_size; (void)ws_size;
    const float* x      = (const float*)d_in[0];
    const float* Aq     = (const float*)d_in[1];
    const float* Bq     = (const float*)d_in[2];
    const float* Av     = (const float*)d_in[3];
    const float* Bv     = (const float*)d_in[4];
    const float* qkv_w  = (const float*)d_in[5];
    const float* qkv_b  = (const float*)d_in[6];
    const float* logits = (const float*)d_in[7];
    float* out = (float*)d_out;

    // workspace layout (all offsets 256B aligned), total ~26.5 MB
    char* ws = (char*)d_ws;
    size_t off = 0;
    float* coef = (float*)(ws + off);            off += 256;
    unsigned short* xh = (unsigned short*)(ws + off); off += (size_t)BNP * DD * 2;
    unsigned short* xl = (unsigned short*)(ws + off); off += (size_t)BNP * DD * 2;
    unsigned short* wh = (unsigned short*)(ws + off); off += (size_t)E3D * DD * 2;
    unsigned short* wl = (unsigned short*)(ws + off); off += (size_t)E3D * DD * 2;

    lora_coef_kernel<<<2 * TT, 256, 0, stream>>>(Aq, Bq, Av, Bv, logits, coef);
    lora_xsplit_kernel<<<(BNP * DD + 255) / 256, 256, 0, stream>>>(x, xh, xl);
    lora_weff_kernel<<<(E3D * DD + 255) / 256, 256, 0, stream>>>(
        qkv_w, Aq, Bq, Av, Bv, coef, wh, wl);

    dim3 grid(BNP / 64, E3D / 256);   // 99 x 9
    lora_qkv_gemm_kernel<<<grid, 256, 0, stream>>>(xh, xl, wh, wl, qkv_b, out);
}
